// AfmoeFlashAttention_33371895890030
// MI455X (gfx1250) — matrix operand order repaired and checked by reading, not compiled, not run on hardware
//
#include <hip/hip_runtime.h>
#include <hip/hip_bf16.h>
#include <stdint.h>

// Problem constants (match reference)
#define T_TOK 8192
#define HID   2048
#define NHEAD 16
#define NKVH  4
#define DHEAD 128
#define SEQS  8
#define SLEN  1024

typedef __attribute__((ext_vector_type(16))) __bf16 bf16x16;
typedef __attribute__((ext_vector_type(8)))  float  floatx8;

static __device__ inline floatx8 wmma_bf16f32(bf16x16 a, bf16x16 b, floatx8 c) {
  // D(16x16,f32) = A(16x32,bf16) * B(32x16,bf16) + C
  return __builtin_amdgcn_wmma_f32_16x16x32_bf16(false, a, false, b, (short)0, c,
                                                 false, false);
}

// ---------------------------------------------------------------------------
// CDNA5 async global->LDS copy (ASYNCcnt-tracked). VDST = LDS byte offset
// (low 32 bits of a generic shared pointer: flat->LDS truncates, ISA 10.2),
// VADDR = 64-bit global address, GV mode (saddr = off). 16 bytes per lane.
static __device__ inline void async_b128(const void* gptr, void* ldsptr) {
  uint32_t loff = (uint32_t)(uintptr_t)ldsptr;
  asm volatile("global_load_async_to_lds_b128 %0, %1, off"
               :
               : "v"(loff), "v"(gptr)
               : "memory");
}
static __device__ inline void wait_async0() {
  asm volatile("s_wait_asynccnt 0x0" ::: "memory");
}

// A-matrix (16x32 bf16) fragment: lane = m + 16*h; vector index i holds
// k = i + 16*h. 16 contiguous elements per lane ->
// compiler emits 2x ds_load_b128 per lane.
static __device__ inline bf16x16 lds_a_frag(const __bf16* base, int stride) {
  const int lane = threadIdx.x & 31;
  const int m = lane & 15;
  const int h = lane >> 4;
  bf16x16 a;
#pragma unroll
  for (int i = 0; i < 16; ++i) {
    int k = i + 16 * h;
    a[i] = base[m * stride + k];
  }
  return a;
}

// B-matrix (32x16) fragment from N x K row-major data (transposed staging):
// lane = n + 16*h; vector index i holds k = i + 16*h -> 16 contiguous
// elements per lane -> 2x ds_load_b128.
static __device__ inline bf16x16 lds_bT_frag(const __bf16* base, int stride) {
  const int lane = threadIdx.x & 31;
  const int n = lane & 15;
  const int h = lane >> 4;
  bf16x16 b;
#pragma unroll
  for (int i = 0; i < 16; ++i) {
    int k = i + 16 * h;
    b[i] = base[n * stride + k];
  }
  return b;
}

// B fragment from K x N row-major data (used for P*V where V is key-major)
static __device__ inline bf16x16 lds_b_frag(const __bf16* base, int stride) {
  const int lane = threadIdx.x & 31;
  const int n = lane & 15;
  const int h = lane >> 4;
  bf16x16 b;
#pragma unroll
  for (int i = 0; i < 16; ++i) {
    int k = i + 16 * h;
    b[i] = base[k * stride + n];
  }
  return b;
}

// ---------------------------------------------------------------------------
__global__ void cvt_f32_bf16(const float* __restrict__ in, __bf16* __restrict__ out,
                             size_t n) {
  size_t i = (size_t)blockIdx.x * blockDim.x + threadIdx.x;
  if (i < n) out[i] = (__bf16)in[i];
}

// ---------------------------------------------------------------------------
// Tiled bf16 GEMM: C[M x N] = A[M x K] * B[K x N]. Block = 256 threads
// (8 waves), 128x128 block tile, K-chunks of 32, double-buffered LDS.
// A staged via async global->LDS b128; B staged transposed (N x K) through
// registers so both fragment types load as contiguous b128.
// Each wave computes 32x64 (2x4 WMMA accumulators).
template <bool OUT_BF16>
__global__ __launch_bounds__(256) void gemm_bf16(const __bf16* __restrict__ A,
                                                 const __bf16* __restrict__ B,
                                                 void* __restrict__ C,
                                                 int N, int K) {
  __shared__ __bf16 sA[2][128 * 32];   // [m][k]
  __shared__ __bf16 sBT[2][128 * 32];  // [n][k] (transposed)
  const int m0 = blockIdx.x * 128;
  const int n0 = blockIdx.y * 128;
  const int tid = threadIdx.x;
  const int wave = tid >> 5;
  const int mo = (wave >> 1) * 32;  // wave row offset within tile
  const int no = (wave & 1) * 64;   // wave col offset within tile

  floatx8 acc[2][4];
#pragma unroll
  for (int i = 0; i < 2; ++i)
#pragma unroll
    for (int j = 0; j < 4; ++j) acc[i][j] = (floatx8){0.f, 0.f, 0.f, 0.f,
                                                      0.f, 0.f, 0.f, 0.f};

  const int arow = tid >> 1, ahalf = tid & 1;   // A: 128 rows x 2 halves
  const int brow = tid >> 3, bseg = tid & 3 ? tid & 7 : tid & 7;  // B: 32 rows x 8 segs

  auto stage = [&](int kc, int buf) {
    // A tile 128x32: 2 async b128 per thread
    const __bf16* ga = A + (size_t)(m0 + arow) * K + kc + ahalf * 16;
    __bf16* la = &sA[buf][arow * 32 + ahalf * 16];
    async_b128(ga, la);
    async_b128(ga + 8, la + 8);
    // B tile 32x128 -> transpose into sBT[n][k]
    const uint4* gb = (const uint4*)(B + (size_t)(kc + brow) * N + n0 + bseg * 16);
    uint4 t[2];
    t[0] = gb[0];
    t[1] = gb[1];
    const __bf16* e = (const __bf16*)t;
#pragma unroll
    for (int j = 0; j < 16; ++j) sBT[buf][(bseg * 16 + j) * 32 + brow] = e[j];
  };

  const int nk = K / 32;
  stage(0, 0);
  for (int c = 0; c < nk; ++c) {
    const int cur = c & 1;
    wait_async0();
    __syncthreads();
    if (c + 1 < nk) stage((c + 1) * 32, cur ^ 1);

    bf16x16 a0 = lds_a_frag(&sA[cur][mo * 32], 32);
    bf16x16 a1 = lds_a_frag(&sA[cur][(mo + 16) * 32], 32);
#pragma unroll
    for (int j = 0; j < 4; ++j) {
      bf16x16 bt = lds_bT_frag(&sBT[cur][(no + j * 16) * 32], 32);
      acc[0][j] = wmma_bf16f32(a0, bt, acc[0][j]);
      acc[1][j] = wmma_bf16f32(a1, bt, acc[1][j]);
    }
  }

  // Store: C/D layout -> row m = r + 8*(lane>=16), col n = lane&15
  const int lane = tid & 31;
  const int cn = lane & 15;
  const int mh = lane >> 4;
#pragma unroll
  for (int ti = 0; ti < 2; ++ti)
#pragma unroll
    for (int tj = 0; tj < 4; ++tj)
#pragma unroll
      for (int r = 0; r < 8; ++r) {
        int m = m0 + mo + ti * 16 + r + 8 * mh;
        int n = n0 + no + tj * 16 + cn;
        if (OUT_BF16)
          ((__bf16*)C)[(size_t)m * N + n] = (__bf16)acc[ti][tj][r];
        else
          ((float*)C)[(size_t)m * N + n] = acc[ti][tj][r];
      }
}

// ---------------------------------------------------------------------------
// Per-(token, head) RMSNorm + RoPE, fp32 in -> bf16 out.
__global__ __launch_bounds__(128) void norm_rope(const float* __restrict__ in,
                                                 const float* __restrict__ w,
                                                 const float* __restrict__ cosb,
                                                 const float* __restrict__ sinb,
                                                 __bf16* __restrict__ out, int NH) {
  const int t = blockIdx.x;
  const int hh = blockIdx.y;
  const int d = threadIdx.x;  // 0..127
  const float* row = in + ((size_t)t * NH + hh) * DHEAD;
  float x = row[d];
  __shared__ float red[DHEAD];
  red[d] = x * x;
  __syncthreads();
  for (int st = 64; st > 0; st >>= 1) {
    if (d < st) red[d] += red[d + st];
    __syncthreads();
  }
  float inv = rsqrtf(red[0] / (float)DHEAD + 1e-6f);
  int pd = (d < 64) ? d + 64 : d - 64;
  float xn = x * inv * w[d];
  float xpn = row[pd] * inv * w[pd];
  float rot = (d < 64) ? -xpn : xpn;
  float c = cosb[(size_t)t * DHEAD + d];
  float s = sinb[(size_t)t * DHEAD + d];
  out[((size_t)t * NH + hh) * DHEAD + d] = (__bf16)(xn * c + rot * s);
}

// ---------------------------------------------------------------------------
// Causal GQA flash attention. One wave per (seq, head, 16-row Q tile).
// Async-staged, double-buffered 32-key K/V chunks; online softmax;
// QK^T and P*V via WMMA bf16.
__global__ __launch_bounds__(32) void flash_attn(const __bf16* __restrict__ Qb,
                                                 const __bf16* __restrict__ Kb,
                                                 const __bf16* __restrict__ Vb,
                                                 float* __restrict__ outO) {
  const int qt = blockIdx.x;  // Q tile within sequence (0..63)
  const int h = blockIdx.y;   // head (0..15)
  const int sq = blockIdx.z;  // sequence (0..7)
  const int kvh = h >> 2;     // GQA: head h -> kv head h/4
  const int lane = threadIdx.x;
  const int n = lane & 15;
  const int mh = lane >> 4;

  __shared__ __bf16 sQ[16 * 128];
  __shared__ __bf16 sK[2][32 * 128];
  __shared__ __bf16 sV[2][32 * 128];
  __shared__ __bf16 sP[16 * 32];

  {  // async stage Q tile (16 x 128): 128B (8 x b128) per lane
    int row = lane >> 1;
    int ho = (lane & 1) * 64;  // elements
    const __bf16* g =
        Qb + (((size_t)(sq * SLEN + qt * 16 + row)) * NHEAD + h) * DHEAD + ho;
    __bf16* l = sQ + row * 128 + ho;
#pragma unroll
    for (int i = 0; i < 8; ++i) async_b128(g + i * 8, l + i * 8);
  }

  auto stage_kv = [&](int ch, int buf) {
    // one 128-wide key row per lane, 16 x b128 each for K and V
    size_t kb = (((size_t)(sq * SLEN + ch * 32 + lane)) * NKVH + kvh) * DHEAD;
    const __bf16* gk = Kb + kb;
    const __bf16* gv = Vb + kb;
    __bf16* lk = &sK[buf][lane * 128];
    __bf16* lv = &sV[buf][lane * 128];
#pragma unroll
    for (int i = 0; i < 16; ++i) {
      async_b128(gk + i * 8, lk + i * 8);
      async_b128(gv + i * 8, lv + i * 8);
    }
  };

  floatx8 o[8] = {};  // 16x128 f32 output as 8 C-fragments
  float mi[8], li[8];
#pragma unroll
  for (int r = 0; r < 8; ++r) { mi[r] = -1e30f; li[r] = 0.f; }

  const int nchunks = (qt * 16 + 16 + 31) / 32;  // keys needed = qt*16+16
  const float scale = 0.08838834764831845f;      // 1/sqrt(128)

  stage_kv(0, 0);
  for (int ch = 0; ch < nchunks; ++ch) {
    const int cur = ch & 1;
    wait_async0();
    __syncthreads();
    if (ch + 1 < nchunks) stage_kv(ch + 1, cur ^ 1);

    // S = Q * K^T for 2 key sub-tiles of 16, K-dim 128 in 4 WMMA each
    floatx8 s0 = {}, s1 = {};
#pragma unroll
    for (int kc = 0; kc < 128; kc += 32) {
      bf16x16 a = lds_a_frag(sQ + kc, 128);
      bf16x16 b0 = lds_bT_frag(&sK[cur][kc], 128);
      bf16x16 b1 = lds_bT_frag(&sK[cur][16 * 128 + kc], 128);
      s0 = wmma_bf16f32(a, b0, s0);
      s1 = wmma_bf16f32(a, b1, s1);
    }

    // Online softmax in C/D layout: row m = r + 8*mh, col = n (+16 for s1)
    float fac[8], p0[8], p1[8];
#pragma unroll
    for (int r = 0; r < 8; ++r) {
      int m = r + 8 * mh;
      int qg = qt * 16 + m;
      int kg0 = ch * 32 + n;
      int kg1 = kg0 + 16;
      float v0 = (kg0 <= qg) ? s0[r] * scale : -1e30f;
      float v1 = (kg1 <= qg) ? s1[r] * scale : -1e30f;
      float mx = fmaxf(v0, v1);
#pragma unroll
      for (int x = 1; x < 16; x <<= 1) mx = fmaxf(mx, __shfl_xor(mx, x, 32));
      float mnew = fmaxf(mi[r], mx);
      fac[r] = __expf(mi[r] - mnew);
      p0[r] = __expf(v0 - mnew);
      p1[r] = __expf(v1 - mnew);
      float rs = p0[r] + p1[r];
#pragma unroll
      for (int x = 1; x < 16; x <<= 1) rs += __shfl_xor(rs, x, 32);
      li[r] = li[r] * fac[r] + rs;
      mi[r] = mnew;
    }

    // Restage P (16x32) in LDS row-major for the A-fragment of P*V
#pragma unroll
    for (int r = 0; r < 8; ++r) {
      int m = r + 8 * mh;
      sP[m * 32 + n] = (__bf16)p0[r];
      sP[m * 32 + 16 + n] = (__bf16)p1[r];
    }
    __syncthreads();

    // Rescale running output, then O += P * V (8 WMMA across DH=128)
#pragma unroll
    for (int j = 0; j < 8; ++j)
#pragma unroll
      for (int r = 0; r < 8; ++r) o[j][r] *= fac[r];

    bf16x16 pa = lds_a_frag(sP, 32);
#pragma unroll
    for (int j = 0; j < 8; ++j) {
      bf16x16 vb = lds_b_frag(&sV[cur][j * 16], 128);
      o[j] = wmma_bf16f32(pa, vb, o[j]);
    }
  }

  // Normalize rows and write attention output fp32 [t][h][dh]
#pragma unroll
  for (int j = 0; j < 8; ++j)
#pragma unroll
    for (int r = 0; r < 8; ++r) {
      int m = r + 8 * mh;
      float val = o[j][r] / li[r];
      size_t idx =
          (((size_t)(sq * SLEN + qt * 16 + m)) * NHEAD + h) * DHEAD + j * 16 + n;
      outO[idx] = val;
    }
}

// ---------------------------------------------------------------------------
// attn * sigmoid(gate) -> bf16 for the output projection
__global__ void gate_mul(const float* __restrict__ attn, const float* __restrict__ g,
                         __bf16* __restrict__ out, size_t nelem) {
  size_t i = (size_t)blockIdx.x * blockDim.x + threadIdx.x;
  if (i < nelem) {
    float sig = 1.f / (1.f + __expf(-g[i]));
    out[i] = (__bf16)(attn[i] * sig);
  }
}

// ---------------------------------------------------------------------------
extern "C" void kernel_launch(void* const* d_in, const int* in_sizes, int n_in,
                              void* d_out, int out_size, void* d_ws, size_t ws_size,
                              hipStream_t stream) {
  (void)in_sizes; (void)n_in; (void)out_size; (void)ws_size;
  const float* x    = (const float*)d_in[0];
  const float* cosb = (const float*)d_in[1];
  const float* sinb = (const float*)d_in[2];
  const float* Wq   = (const float*)d_in[3];
  const float* Wk   = (const float*)d_in[4];
  const float* Wv   = (const float*)d_in[5];
  const float* Wg   = (const float*)d_in[6];
  const float* Wo   = (const float*)d_in[7];
  const float* qnw  = (const float*)d_in[8];
  const float* knw  = (const float*)d_in[9];
  // d_in[10] = cu_seqlens: fixed L=1024 segments, baked into the grid.
  float* out = (float*)d_out;

  char* ws = (char*)d_ws;
  size_t off = 0;
  auto alloc = [&](size_t bytes) {
    char* p = ws + off;
    off = (off + bytes + 255) & ~(size_t)255;
    return p;
  };

  __bf16* Xb  = (__bf16*)alloc((size_t)T_TOK * HID * 2);
  __bf16* Wqb = (__bf16*)alloc((size_t)HID * 2048 * 2);
  __bf16* Wkb = (__bf16*)alloc((size_t)HID * 512 * 2);
  __bf16* Wvb = (__bf16*)alloc((size_t)HID * 512 * 2);
  __bf16* Wgb = (__bf16*)alloc((size_t)HID * 2048 * 2);
  __bf16* Wob = (__bf16*)alloc((size_t)HID * 2048 * 2);
  float*  Qf  = (float*)alloc((size_t)T_TOK * 2048 * 4);
  float*  Kf  = (float*)alloc((size_t)T_TOK * 512 * 4);
  float*  Gf  = (float*)alloc((size_t)T_TOK * 2048 * 4);
  __bf16* Qb  = (__bf16*)alloc((size_t)T_TOK * 2048 * 2);
  __bf16* Kb  = (__bf16*)alloc((size_t)T_TOK * 512 * 2);
  __bf16* Vb  = (__bf16*)alloc((size_t)T_TOK * 512 * 2);
  float*  Af  = (float*)alloc((size_t)T_TOK * 2048 * 4);
  __bf16* Ab  = (__bf16*)alloc((size_t)T_TOK * 2048 * 2);

  auto cvt = [&](const float* src, __bf16* dst, size_t n) {
    cvt_f32_bf16<<<dim3((unsigned)((n + 255) / 256)), dim3(256), 0, stream>>>(src, dst, n);
  };
  cvt(x,  Xb,  (size_t)T_TOK * HID);
  cvt(Wq, Wqb, (size_t)HID * 2048);
  cvt(Wk, Wkb, (size_t)HID * 512);
  cvt(Wv, Wvb, (size_t)HID * 512);
  cvt(Wg, Wgb, (size_t)HID * 2048);
  cvt(Wo, Wob, (size_t)HID * 2048);

  // Projections (M=8192 tokens; 128x128 block tiles)
  gemm_bf16<false><<<dim3(T_TOK / 128, 2048 / 128), dim3(256), 0, stream>>>(Xb, Wqb, Qf, 2048, HID);
  gemm_bf16<false><<<dim3(T_TOK / 128, 512 / 128),  dim3(256), 0, stream>>>(Xb, Wkb, Kf, 512,  HID);
  gemm_bf16<true ><<<dim3(T_TOK / 128, 512 / 128),  dim3(256), 0, stream>>>(Xb, Wvb, Vb, 512,  HID);
  gemm_bf16<false><<<dim3(T_TOK / 128, 2048 / 128), dim3(256), 0, stream>>>(Xb, Wgb, Gf, 2048, HID);

  // QK norm + RoPE -> bf16
  norm_rope<<<dim3(T_TOK, NHEAD), dim3(128), 0, stream>>>(Qf, qnw, cosb, sinb, Qb, NHEAD);
  norm_rope<<<dim3(T_TOK, NKVH),  dim3(128), 0, stream>>>(Kf, knw, cosb, sinb, Kb, NKVH);

  // Flash attention
  flash_attn<<<dim3(SLEN / 16, NHEAD, SEQS), dim3(32), 0, stream>>>(Qb, Kb, Vb, Af);

  // Gate + output projection
  gate_mul<<<dim3((unsigned)(((size_t)T_TOK * 2048 + 255) / 256)), dim3(256), 0, stream>>>(
      Af, Gf, Ab, (size_t)T_TOK * 2048);
  gemm_bf16<false><<<dim3(T_TOK / 128, 2048 / 128), dim3(256), 0, stream>>>(Ab, Wob, out, 2048, HID);
}
